// DeepMPDRModel_72232759984525
// MI455X (gfx1250) — matrix operand order repaired and checked by reading, not compiled, not run on hardware
//
#include <hip/hip_runtime.h>
#include <hip/hip_bf16.h>
#include <math.h>

// ---------------------------------------------------------------------------
// DeepMPDR on MI455X (gfx1250), wave32 + WMMA f16 (f32 accumulate).
//
//   t  = sw @ P^T          (P = phi1@phi2, precomputed)   [16 x 192] x [192 x 192]
//   b  = s  @ M^T          (M[j,k] = phi4[j,k]*phi5[k,j]) [16 x 192] x [192 x 192]
//   w  = v  @ phi3^T                                       [16 x 64 ] x [64  x 192]
//   out = t*b ; global min/max (2-pass via ws) ; gate ; sqrt(r^2+i^2)
//
// Memory-bound at ~365MB of HBM traffic (~16us @ 23.3TB/s); f16 WMMA keeps
// compute far below the memory roof. LDS held at 158.7KB (<=160KB) so two
// workgroups can co-reside per 320KB WGP. Fragment image staged into LDS by
// the Tensor Data Mover (one tensor_load_to_lds per block, TENSORcnt-tracked).
// ---------------------------------------------------------------------------

#define N_GRID 181
#define N_ANT  64
#define BATCH  65536
#define NGP    192          // padded grid dim (12 tiles of 16)
#define NTN    12           // N tiles
#define KT_T   6            // K tiles of 32 over 192
#define ROW_TILES (BATCH / 16)

typedef __attribute__((ext_vector_type(16))) _Float16 v16h;
typedef __attribute__((ext_vector_type(8)))  _Float16 v8h;
typedef __attribute__((ext_vector_type(8)))  float    v8f;
typedef __attribute__((ext_vector_type(4)))  unsigned int u32x4;
typedef __attribute__((ext_vector_type(8)))  int          i32x8;
typedef __attribute__((ext_vector_type(4)))  int          i32x4;

// ----- workspace byte offsets ----------------------------------------------
#define WS_FRAG_P3   0u          // 24 frags (phi3^T)  : 24576 B
#define WS_FRAG_PT   24576u      // 72 frags (P^T)     : 73728 B  (LDS-resident with P3)
#define WS_FRAG_MT   98304u      // 72 frags (M^T)     : 73728 B  (streamed from L2)
#define WS_P         262144u     // f32 [181*181] temp
#define WS_SCAL      458752u     // 4 x u32 order-preserving float keys
#define WS_OUTR      (1u << 20)  // f32 [B*181]
#define WS_OUTI      (50u << 20) // f32 [B*181]

// ----- helpers -------------------------------------------------------------
__device__ __forceinline__ void waitlds() {
  asm volatile("s_wait_dscnt 0" ::: "memory");   // CDNA5 split DS counter wait
}

__device__ __forceinline__ v16h join8(v8h a, v8h b) {
  return __builtin_shufflevector(a, b, 0,1,2,3,4,5,6,7,8,9,10,11,12,13,14,15);
}

// B fragment: 512 f16 = two 512B chunks; lane L reads 16B at L*16 (+512).
__device__ __forceinline__ v16h load_bfrag(const _Float16* fb, int lane) {
  v8h a = *(const v8h*)(fb + lane * 8);
  v8h b = *(const v8h*)(fb + 256 + lane * 8);
  return join8(a, b);
}

// A fragment (16xK f16): lane<16 -> K {k0..k0+15},
// lane>=16 -> K {k0+16..k0+31}; row M = lane&15.
__device__ __forceinline__ v16h build_afrag(const _Float16* st, int k0, int stride, int lane) {
  int m   = lane & 15;
  int off = k0 + ((lane >> 4) << 4);
  const _Float16* p = st + m * stride + off;
  v8h a = *(const v8h*)(p);
  v8h b = *(const v8h*)(p + 8);
  return join8(a, b);
}

// order-preserving float<->u32 keys for atomic min/max
__device__ __forceinline__ unsigned fkey(float f) {
  unsigned b = __float_as_uint(f);
  return (b & 0x80000000u) ? ~b : (b | 0x80000000u);
}
__device__ __forceinline__ float key2f(unsigned k) {
  unsigned b = (k & 0x80000000u) ? (k ^ 0x80000000u) : ~k;
  return __uint_as_float(b);
}

// ----- kernel 1: P = phi1 @ phi2 -------------------------------------------
__global__ void k_prep_P(const float* __restrict__ phi1, const float* __restrict__ phi2,
                         float* __restrict__ P) {
  int idx = blockIdx.x * blockDim.x + threadIdx.x;
  if (idx >= N_GRID * N_GRID) return;
  int n = idx / N_GRID, k = idx - n * N_GRID;
  float acc = 0.f;
  #pragma unroll 4
  for (int a = 0; a < N_ANT; ++a) acc += phi1[n * N_ANT + a] * phi2[a * N_GRID + k];
  P[idx] = acc;                       // P[n][k]
}

// ----- kernel 2: build f16 B-fragment images + reset min/max keys ----------
__global__ void k_prep_frags(const float* __restrict__ phi3, const float* __restrict__ P,
                             const float* __restrict__ phi4, const float* __restrict__ phi5,
                             _Float16* __restrict__ frag_img, unsigned* __restrict__ scal) {
  int t = blockIdx.x * blockDim.x + threadIdx.x;
  if (t < 4) scal[t] = (t & 1) ? 0u : 0xFFFFFFFFu;   // [min,max] x {r,i}
  if (t >= 168 * 512) return;
  int frag = t >> 9;
  int within = t & 511;
  int lane, e;
  if (within < 256) { lane = within >> 3; e = within & 7; }
  else              { lane = (within - 256) >> 3; e = 8 + ((within - 256) & 7); }
  int nlo  = lane & 15;
  int koff = (lane < 16) ? 0 : 16;     // B 32x16: lanes 0-15 K=0..15, 16-31 K=16..31
  float val = 0.f;
  if (frag < 24) {                      // phi3^T[K][N] = phi3[N][K], K<64 always
    int kt = frag / NTN, nt = frag % NTN;
    int K = kt * 32 + koff + e, N = nt * 16 + nlo;
    if (N < N_GRID) val = phi3[N * N_ANT + K];
  } else if (frag < 96) {               // P^T[K][N] = P[N][K]
    int f = frag - 24;
    int kt = f / NTN, nt = f % NTN;
    int K = kt * 32 + koff + e, N = nt * 16 + nlo;
    if (K < N_GRID && N < N_GRID) val = P[N * N_GRID + K];
  } else {                              // M^T[K][N] = phi4[N][K] * phi5[K][N]
    int f = frag - 96;
    int kt = f / NTN, nt = f % NTN;
    int K = kt * 32 + koff + e, N = nt * 16 + nlo;
    if (K < N_GRID && N < N_GRID) val = phi4[N * N_GRID + K] * phi5[K * N_GRID + N];
  }
  frag_img[t] = (_Float16)val;
}

// ----- kernel 3: main WMMA pipeline ----------------------------------------
__global__ __launch_bounds__(128) void k_main(const float* __restrict__ s_minus,
                                              const float* __restrict__ bf,
                                              unsigned char* __restrict__ ws) {
  // 98304 + 25600 + 25600 + 9216 = 158720 B static LDS (<= 160KB: 2 WGs/WGP)
  __shared__ __align__(16) _Float16 lds_frag[96 * 512];   // phi3^T + P^T fragments
  __shared__ __align__(16) _Float16 s_st[4][16 * 200];    // per-wave staging (stride 200)
  __shared__ __align__(16) _Float16 w_st[4][16 * 200];
  __shared__ __align__(16) _Float16 v_st[4][16 * 72];

  // ---- stage phi3^T + P^T fragment image into LDS -------------------------
#if defined(__gfx1250__) && __has_builtin(__builtin_amdgcn_tensor_load_to_lds) && \
    __has_builtin(__builtin_amdgcn_s_wait_tensorcnt)
  if (threadIdx.x < 32) {               // one wave issues the TDM op (EXEC ignored)
    unsigned long long ga = (unsigned long long)(ws + WS_FRAG_P3);
    unsigned lds_off = (unsigned)(unsigned long long)(void*)lds_frag; // LDS aperture: addr[31:0]
    // D# group 0: count=1 | lds_addr | global_addr | type=2  (ISA 8.3)
    u32x4 g0;
    g0[0] = 1u;                                        // count=1, user desc
    g0[1] = lds_off;                                   // lds_addr[31:0]
    g0[2] = (unsigned)(ga & 0xFFFFFFFFu);              // global_addr[31:0]
    g0[3] = (unsigned)((ga >> 32) & 0x1FFFFFFu) | (2u << 30);  // addr[56:32] | type=2
    // D# group 1: data_size=8B; 96KB as one 12288-unit row (ISA 8.4)
    const int UNITS = (96 * 1024) / 8;                 // 12288
    i32x8 g1;
    g1[0] = 3 << 16;                                   // data_size=3 (8B), no mask/flags
    g1[1] = (UNITS & 0xFFFF) << 16;                    // tensor_dim0[15:0]
    g1[2] = 1 << 16;                                   // tensor_dim0 hi=0 | tensor_dim1=1
    g1[3] = (UNITS & 0xFFFF) << 16;                    // tensor_dim1 hi=0 | tile_dim0
    g1[4] = 1;                                         // tile_dim1=1 | tile_dim2=0
    g1[5] = UNITS;                                     // tensor_dim0_stride[31:0]
    g1[6] = 0;                                         // stride hi | dim1_stride lo
    g1[7] = 0;
    i32x4 gz4 = {0, 0, 0, 0};                          // groups 2/3 unused (<=2D)
    i32x8 gz8 = {0, 0, 0, 0, 0, 0, 0, 0};              // extra group (clang-23 6-arg form)
    __builtin_amdgcn_tensor_load_to_lds(g0, g1, gz4, gz4, gz8, 0);
    __builtin_amdgcn_s_wait_tensorcnt(0);
  }
#else
  {
    const uint4* src = (const uint4*)(ws + WS_FRAG_P3);
    uint4* dst = (uint4*)lds_frag;
    for (int i = threadIdx.x; i < (96 * 1024) / 16; i += blockDim.x) dst[i] = src[i];
  }
#endif
  // warm L2 lines for M^T fragments (global_prefetch_b8)
  for (unsigned off = threadIdx.x * 256u; off < 72u * 1024u; off += blockDim.x * 256u)
    __builtin_prefetch(ws + WS_FRAG_MT + off, 0, 3);
  __syncthreads();

  const _Float16* mt_g = (const _Float16*)(ws + WS_FRAG_MT);
  float* outr = (float*)(ws + WS_OUTR);
  float* outi = (float*)(ws + WS_OUTI);

  const int lane = threadIdx.x & 31;
  const int wv   = threadIdx.x >> 5;
  const int gw   = blockIdx.x * 4 + wv;
  const int nw   = gridDim.x * 4;
  const int l15  = lane & 15;
  const int hi8  = (lane >> 4) << 3;            // C layout: lanes>=16 hold rows M+8
  _Float16* sst = s_st[wv];
  _Float16* wst = w_st[wv];
  _Float16* vst = v_st[wv];

  unsigned mnk[2] = {0xFFFFFFFFu, 0xFFFFFFFFu};
  unsigned mxk[2] = {0u, 0u};

  for (int rt = gw; rt < ROW_TILES; rt += nw) {
    const int row0 = rt << 4;
    for (int h = 0; h < 2; ++h) {
      // ---- stage s (f16, zero-padded to 192 cols) and v into LDS ----------
      for (int i = lane; i < 16 * NGP; i += 32) {
        int r = i / NGP, c = i - r * NGP;
        float sv = (c < N_GRID) ? s_minus[(size_t)(row0 + r) * (2 * N_GRID) + h * N_GRID + c]
                                : 0.f;
        sst[r * 200 + c] = (_Float16)sv;
      }
      for (int i = lane; i < 16 * N_ANT; i += 32) {
        int r = i >> 6, c = i & 63;
        vst[r * 72 + c] = (_Float16)bf[(size_t)(row0 + r) * (2 * N_ANT) + h * N_ANT + c];
      }
      waitlds();

      // ---- stage 1: w = v @ phi3^T  (K=64 -> 2 k-steps) -------------------
      v16h av0 = build_afrag(vst, 0, 72, lane);
      v16h av1 = build_afrag(vst, 32, 72, lane);
      for (int nt = 0; nt < NTN; ++nt) {
        v8f acc = {};
        acc = __builtin_amdgcn_wmma_f32_16x16x32_f16(false, av0, false,
                load_bfrag(lds_frag + (0 * NTN + nt) * 512, lane), (short)0, acc, false, false);
        acc = __builtin_amdgcn_wmma_f32_16x16x32_f16(false, av1, false,
                load_bfrag(lds_frag + (1 * NTN + nt) * 512, lane), (short)0, acc, false, false);
        int c0 = nt * 16 + l15;
        #pragma unroll
        for (int r = 0; r < 8; ++r) wst[(hi8 + r) * 200 + c0] = (_Float16)acc[r];
      }
      waitlds();

      // ---- A fragments: sw = s*w (packed f16 mul) and s -------------------
      v16h a_sw[KT_T], a_s[KT_T];
      #pragma unroll
      for (int kt = 0; kt < KT_T; ++kt) {
        v16h sf = build_afrag(sst, kt * 32, 200, lane);
        v16h wf = build_afrag(wst, kt * 32, 200, lane);
        a_s[kt]  = sf;
        a_sw[kt] = sf * wf;                     // v_pk_mul_f16
      }

      // ---- stage 2+3: t = sw @ P^T , b = s @ M^T , out = t*b --------------
      for (int nt = 0; nt < NTN; ++nt) {
        v8f tacc = {}, bacc = {};
        #pragma unroll
        for (int kt = 0; kt < KT_T; ++kt)
          tacc = __builtin_amdgcn_wmma_f32_16x16x32_f16(false, a_sw[kt], false,
                   load_bfrag(lds_frag + (24 + kt * NTN + nt) * 512, lane),
                   (short)0, tacc, false, false);
        #pragma unroll
        for (int kt = 0; kt < KT_T; ++kt)
          bacc = __builtin_amdgcn_wmma_f32_16x16x32_f16(false, a_s[kt], false,
                   load_bfrag(mt_g + (kt * NTN + nt) * 512, lane),
                   (short)0, bacc, false, false);
        int col = nt * 16 + l15;
        if (col < N_GRID) {
          float* outp = h ? outi : outr;
          #pragma unroll
          for (int r = 0; r < 8; ++r) {
            float o = tacc[r] * bacc[r];
            unsigned k = fkey(o);
            mnk[h] = min(mnk[h], k);
            mxk[h] = max(mxk[h], k);
            outp[(size_t)(row0 + hi8 + r) * N_GRID + col] = o;
          }
        }
      }
    }
  }

  // ---- wave-level min/max reduce, then one atomic per wave per key --------
  unsigned* scal = (unsigned*)(ws + WS_SCAL);
  #pragma unroll
  for (int h = 0; h < 2; ++h) {
    unsigned mn = mnk[h], mx = mxk[h];
    for (int off = 16; off > 0; off >>= 1) {
      mn = min(mn, (unsigned)__shfl_xor((int)mn, off, 32));
      mx = max(mx, (unsigned)__shfl_xor((int)mx, off, 32));
    }
    if (lane == 0) {
      atomicMin(&scal[2 * h + 0], mn);
      atomicMax(&scal[2 * h + 1], mx);
    }
  }
}

// ----- kernel 4: normalize + gate + complex magnitude ----------------------
__global__ void k_final(const unsigned char* __restrict__ ws,
                        const float* __restrict__ alpha_p, const float* __restrict__ beta_p,
                        float* __restrict__ out, int n) {
  int idx = blockIdx.x * blockDim.x + threadIdx.x;
  if (idx >= n) return;
  const float* outr = (const float*)(ws + WS_OUTR);
  const float* outi = (const float*)(ws + WS_OUTI);
  const unsigned* scal = (const unsigned*)(ws + WS_SCAL);
  float alpha = alpha_p[0], beta = beta_p[0];
  float mnr = key2f(scal[0]), mxr = key2f(scal[1]);
  float mni = key2f(scal[2]), mxi = key2f(scal[3]);
  float r = (outr[idx] - mnr) / (mxr - mnr);
  float i = (outi[idx] - mni) / (mxi - mni);
  float gr = r * (1.f / (1.f + __expf(-(alpha * (r - beta)))));
  float gi = i * (1.f / (1.f + __expf(-(alpha * (i - beta)))));
  out[idx] = sqrtf(gr * gr + gi * gi);
}

// ---------------------------------------------------------------------------
extern "C" void kernel_launch(void* const* d_in, const int* in_sizes, int n_in,
                              void* d_out, int out_size, void* d_ws, size_t ws_size,
                              hipStream_t stream) {
  (void)in_sizes; (void)n_in; (void)out_size; (void)ws_size;
  const float* s_minus = (const float*)d_in[0];
  const float* bf      = (const float*)d_in[1];
  const float* phi1    = (const float*)d_in[2];
  const float* phi2    = (const float*)d_in[3];
  const float* phi3    = (const float*)d_in[4];
  const float* phi4    = (const float*)d_in[5];
  const float* phi5    = (const float*)d_in[6];
  const float* alpha   = (const float*)d_in[7];
  const float* beta    = (const float*)d_in[8];
  unsigned char* ws = (unsigned char*)d_ws;

  float*     P    = (float*)(ws + WS_P);
  _Float16*  frag = (_Float16*)(ws + WS_FRAG_P3);
  unsigned*  scal = (unsigned*)(ws + WS_SCAL);

  k_prep_P<<<(N_GRID * N_GRID + 255) / 256, 256, 0, stream>>>(phi1, phi2, P);
  k_prep_frags<<<(168 * 512 + 255) / 256, 256, 0, stream>>>(phi3, P, phi4, phi5, frag, scal);
  k_main<<<512, 128, 0, stream>>>(s_minus, bf, ws);
  const int n = BATCH * N_GRID;
  k_final<<<(n + 255) / 256, 256, 0, stream>>>(ws, alpha, beta, (float*)d_out, n);
}
